// guided_SE_CAM_Module_50646254355145
// MI455X (gfx1250) — matrix operand order repaired and checked by reading, not compiled, not run on hardware
//
#include <hip/hip_runtime.h>
#include <hip/hip_bf16.h>

// ---------------------------------------------------------------------------
// Problem dims (fixed by the reference)
// ---------------------------------------------------------------------------
#define B_   4
#define C_   2560
#define HW_  4096          // H*W = 64*64
#define Q_   512
#define O_   512
#define SEH_ 160           // C/16
#define EPS_ 1e-5f

typedef __bf16 bf16;
typedef __attribute__((ext_vector_type(8)))  __bf16 v8bf;
typedef __attribute__((ext_vector_type(16))) __bf16 v16bf;
typedef __attribute__((ext_vector_type(8)))  float  v8f;

// staged tile geometry: 128 rows x 32 K-elements (64B) padded to 80B/row so
// 16-lane fragment reads (row stride 20 dwords) hit 16 distinct LDS banks.
#define PADB   80
#define TILEB  (128 * PADB)   // 10240 B per tile

// one 16B per-lane async copy: LDS[ldsaddr] = MEM[gaddr]
__device__ __forceinline__ void async_ld16(unsigned ldsaddr, const void* gaddr)
{
    asm volatile("global_load_async_to_lds_b128 %0, %1, off"
                 :: "v"(ldsaddr), "v"(gaddr) : "memory");
}

// stage one 128x32 bf16 tile (rows row0.., K k0..k0+31) into LDS (2 asyncs/thread)
__device__ __forceinline__ void stage_tile(const bf16* __restrict__ gbase, int ld,
                                           int row0, int k0, char* sdst, int t)
{
    const int r  = t >> 1;           // 0..127
    const int sb = (t & 1) * 32;     // byte offset within the 64B row chunk
    const char* g = (const char*)(gbase + (size_t)(row0 + r) * ld + k0) + sb;
    unsigned l = (unsigned)(size_t)(sdst + r * PADB + sb);
    async_ld16(l,      g);
    async_ld16(l + 16, g + 16);
    __builtin_prefetch(g + 128, 0, 1);   // warm L2 ~2 K-steps ahead of the async engine
}

// ---------------------------------------------------------------------------
// Generic batched bf16 GEMM, f32 accumulation, V_WMMA_F32_16X16X32_BF16.
//   D[M,N] (f32, row-major, ldd) = A[M,K] (bf16 row-major, lda)
//                                x B[K,N] given as BT[N,K] (bf16 row-major, ldb)
// Block: 256 threads = 8 waves; block tile 128x128; wave tile 32x64 (2x4 WMMA).
// Operands are double-buffer staged to LDS with GLOBAL_LOAD_ASYNC_TO_LDS_B128
// (ASYNCcnt) and consumed with DS_LOAD_B128.
// Requires M%128==0, N%128==0, K%64==0 (true for all 4 GEMMs here).
// ---------------------------------------------------------------------------
__global__ __launch_bounds__(256) void k_gemm_bf16(
    const bf16* __restrict__ A,  int lda, long long strideA,
    const bf16* __restrict__ BT, int ldb, long long strideB,
    float*      __restrict__ D,  int ldd, long long strideD,
    int M, int N, int K)
{
    __shared__ __align__(16) char sA[2][TILEB];
    __shared__ __align__(16) char sB[2][TILEB];

    const int batch = blockIdx.z;
    A  += (size_t)batch * strideA;
    BT += (size_t)batch * strideB;
    D  += (size_t)batch * strideD;

    const int t     = threadIdx.x;
    const int lane  = t & 31;
    const int wave  = t >> 5;                // 0..7
    const int wm    = wave >> 1;             // 0..3  -> 4 row groups of 32
    const int wn    = wave & 1;              // 0..1  -> 2 col groups of 64
    const int rowB  = blockIdx.y * 128;      // block row base
    const int colB  = blockIdx.x * 128;      // block col base

    const int mlane = lane & 15;
    const int kh16  = (lane >> 4) * 16;      // A: byte offset of lane's K-half
    const int kb32  = (lane >> 4) * 16;      // B: byte offset of lane's K-half

    v8f acc[2][4] = {};                      // 2 M-tiles x 4 N-tiles

    // per-wave LDS fragment bases (within a tile buffer)
    const int arow[2] = { (wm * 32 +  0 + mlane) * PADB,
                          (wm * 32 + 16 + mlane) * PADB };
    int brow[4];
#pragma unroll
    for (int j = 0; j < 4; ++j) brow[j] = (wn * 64 + j * 16 + mlane) * PADB;

    // prologue: stage buffers 0 and 1 (4 async b128 per thread per buffer)
    stage_tile(A,  lda, rowB, 0,  sA[0], t);
    stage_tile(BT, ldb, colB, 0,  sB[0], t);
    stage_tile(A,  lda, rowB, 32, sA[1], t);
    stage_tile(BT, ldb, colB, 32, sB[1], t);

    int bi = 0;
    for (int k0 = 0; k0 < K; k0 += 32, bi ^= 1) {
        // steady state: allow the next buffer's 4 asyncs to stay in flight
        if (k0 + 32 < K) asm volatile("s_wait_asynccnt 4" ::: "memory");
        else             asm volatile("s_wait_asynccnt 0" ::: "memory");
        __syncthreads();                     // all waves' staging visible

        const char* a = sA[bi];
        const char* b = sB[bi];
        v16bf af[2], bfr[4];
#pragma unroll
        for (int i = 0; i < 2; ++i) {        // A frag: 16B runs at +0 and +32
            v8bf lo = *(const v8bf*)(a + arow[i] + kh16);
            v8bf hi = *(const v8bf*)(a + arow[i] + kh16 + 32);
            af[i] = __builtin_shufflevector(lo, hi, 0,1,2,3,4,5,6,7,8,9,10,11,12,13,14,15);
        }
#pragma unroll
        for (int j = 0; j < 4; ++j) {        // B frag: 16B runs at +0 and +32
            v8bf lo = *(const v8bf*)(b + brow[j] + kb32);
            v8bf hi = *(const v8bf*)(b + brow[j] + kb32 + 32);
            bfr[j] = __builtin_shufflevector(lo, hi, 0,1,2,3,4,5,6,7,8,9,10,11,12,13,14,15);
        }

#pragma unroll
        for (int i = 0; i < 2; ++i)
#pragma unroll
            for (int j = 0; j < 4; ++j)
                acc[i][j] = __builtin_amdgcn_wmma_f32_16x16x32_bf16(
                    false, af[i], false, bfr[j], (short)0, acc[i][j], false, false);

        // WMMAs above consumed the ds_load results (DScnt waited), so after this
        // barrier buffer bi may be overwritten for K-step k0+64.
        __syncthreads();
        if (k0 + 64 < K) {
            stage_tile(A,  lda, rowB, k0 + 64, sA[bi], t);
            stage_tile(BT, ldb, colB, k0 + 64, sB[bi], t);
        }
    }

    // D layout: VGPR e, lanes 0-15 -> (M=e, N=lane); lanes 16-31 -> (M=e+8)
    const int mbase = (lane >> 4) * 8;
    const int ncol  = lane & 15;
#pragma unroll
    for (int i = 0; i < 2; ++i)
#pragma unroll
        for (int j = 0; j < 4; ++j) {
            float* d = D + (size_t)(rowB + wm * 32 + i * 16 + mbase) * ldd
                         + colB + wn * 64 + j * 16 + ncol;
#pragma unroll
            for (int e = 0; e < 8; ++e)
                d[(size_t)e * ldd] = acc[i][j][e];
        }
}

// ---------------------------------------------------------------------------
// x (f32 [B,C,N]) -> x_bf16 [B,C,N] and xT_bf16 [B,N,C] (LDS tile transpose)
// ---------------------------------------------------------------------------
__global__ void k_cvt_transpose(const float* __restrict__ x,
                                bf16* __restrict__ xb, bf16* __restrict__ xtb)
{
    __shared__ bf16 tile[32][33];
    const int b  = blockIdx.z;
    const int c0 = blockIdx.y * 32;
    const int n0 = blockIdx.x * 32;
    const int tx = threadIdx.x;   // 0..31
    const int ty = threadIdx.y;   // 0..7
    const float* xs = x + (size_t)b * C_ * HW_;
#pragma unroll
    for (int i = 0; i < 4; ++i) {
        int  c = c0 + ty + i * 8;
        bf16 h = (bf16)xs[(size_t)c * HW_ + n0 + tx];
        xb[(size_t)b * C_ * HW_ + (size_t)c * HW_ + n0 + tx] = h;
        tile[ty + i * 8][tx] = h;
    }
    __syncthreads();
#pragma unroll
    for (int i = 0; i < 4; ++i) {
        int n = n0 + ty + i * 8;
        xtb[(size_t)b * HW_ * C_ + (size_t)n * C_ + c0 + tx] = tile[tx][ty + i * 8];
    }
}

__global__ void k_cvt(const float* __restrict__ in, bf16* __restrict__ out, int n)
{
    int i = blockIdx.x * blockDim.x + threadIdx.x;
    if (i < n) out[i] = (bf16)in[i];
}

// ---------------------------------------------------------------------------
// pooled[b,c] = mean_n x[b,c,n]
// ---------------------------------------------------------------------------
__global__ void k_pooled(const float* __restrict__ x, float* __restrict__ pooled)
{
    const int row = blockIdx.x;               // b*C + c
    const float* p = x + (size_t)row * HW_;
    float s = 0.f;
    for (int i = threadIdx.x; i < HW_; i += blockDim.x) s += p[i];
    __shared__ float red[256];
    red[threadIdx.x] = s; __syncthreads();
    for (int off = 128; off > 0; off >>= 1) {
        if ((int)threadIdx.x < off) red[threadIdx.x] += red[threadIdx.x + off];
        __syncthreads();
    }
    if (threadIdx.x == 0) pooled[row] = red[0] * (1.0f / HW_);
}

// ---------------------------------------------------------------------------
// SE gate: h1 = relu(pooled @ W1^T + b1); se = sigmoid(h1 @ W2^T + b2)
// ---------------------------------------------------------------------------
__global__ void k_se(const float* __restrict__ pooled,
                     const float* __restrict__ W1, const float* __restrict__ b1,
                     const float* __restrict__ W2, const float* __restrict__ b2,
                     float* __restrict__ se)
{
    const int b = blockIdx.x;
    __shared__ float pl[C_];
    __shared__ float h1[SEH_];
    for (int i = threadIdx.x; i < C_; i += blockDim.x) pl[i] = pooled[b * C_ + i];
    __syncthreads();
    for (int i = threadIdx.x; i < SEH_; i += blockDim.x) {
        float s = b1[i];
        const float* w = W1 + (size_t)i * C_;
        for (int c = 0; c < C_; ++c) s += pl[c] * w[c];
        h1[i] = fmaxf(s, 0.f);
    }
    __syncthreads();
    for (int q = threadIdx.x; q < Q_; q += blockDim.x) {
        float s = b2[q];
        const float* w = W2 + (size_t)q * SEH_;
        for (int i = 0; i < SEH_; ++i) s += h1[i] * w[i];
        se[b * Q_ + q] = 1.f / (1.f + __expf(-s));
    }
}

// ---------------------------------------------------------------------------
// Per-channel batch stats over (B, N): mean and rsqrt(var+eps)
// ---------------------------------------------------------------------------
__global__ void k_bn_stats(const float* __restrict__ y, int Ch,
                           float* __restrict__ mean, float* __restrict__ inv)
{
    const int ch = blockIdx.x;
    float s = 0.f, s2 = 0.f;
    for (int b = 0; b < B_; ++b) {
        const float* p = y + ((size_t)b * Ch + ch) * (size_t)HW_;
        for (int n = threadIdx.x; n < HW_; n += blockDim.x) {
            float v = p[n]; s += v; s2 += v * v;
        }
    }
    __shared__ float r1[256], r2[256];
    r1[threadIdx.x] = s; r2[threadIdx.x] = s2; __syncthreads();
    for (int off = 128; off > 0; off >>= 1) {
        if ((int)threadIdx.x < off) {
            r1[threadIdx.x] += r1[threadIdx.x + off];
            r2[threadIdx.x] += r2[threadIdx.x + off];
        }
        __syncthreads();
    }
    if (threadIdx.x == 0) {
        const float cnt = (float)(B_) * (float)(HW_);
        float m   = r1[0] / cnt;
        float var = r2[0] / cnt - m * m;
        mean[ch] = m;
        inv[ch]  = rsqrtf(var + EPS_);
    }
}

// bn1 + relu applied to bottle_pre -> qf (bf16, energy A-operand)
__global__ void k_bn1_qf(const float* __restrict__ pre,
                         const float* __restrict__ mean, const float* __restrict__ inv,
                         const float* __restrict__ scale, const float* __restrict__ bias,
                         bf16* __restrict__ qf)
{
    const size_t total = (size_t)B_ * Q_ * HW_;
    for (size_t i = blockIdx.x * (size_t)blockDim.x + threadIdx.x; i < total;
         i += (size_t)gridDim.x * blockDim.x) {
        int q = (int)((i / HW_) % Q_);
        float v = (pre[i] - mean[q]) * inv[q] * scale[q] + bias[q];
        qf[i] = (bf16)fmaxf(v, 0.f);
    }
}

// ---------------------------------------------------------------------------
// softmax(max - e) over C  ==  exp(min - e) / sum(exp(min - e))
// one block per (b,q) row
// ---------------------------------------------------------------------------
__global__ void k_softmax(const float* __restrict__ energy, bf16* __restrict__ attn)
{
    const size_t row = blockIdx.x;            // b*Q + q
    const float* e = energy + row * C_;
    __shared__ float red[256];
    float m = 3.4e38f;
    for (int c = threadIdx.x; c < C_; c += blockDim.x) m = fminf(m, e[c]);
    red[threadIdx.x] = m; __syncthreads();
    for (int off = 128; off > 0; off >>= 1) {
        if ((int)threadIdx.x < off) red[threadIdx.x] = fminf(red[threadIdx.x], red[threadIdx.x + off]);
        __syncthreads();
    }
    m = red[0]; __syncthreads();
    float s = 0.f;
    for (int c = threadIdx.x; c < C_; c += blockDim.x) s += __expf(m - e[c]);
    red[threadIdx.x] = s; __syncthreads();
    for (int off = 128; off > 0; off >>= 1) {
        if ((int)threadIdx.x < off) red[threadIdx.x] += red[threadIdx.x + off];
        __syncthreads();
    }
    const float zinv = 1.f / red[0];
    bf16* a = attn + row * C_;
    for (int c = threadIdx.x; c < C_; c += blockDim.x)
        a[c] = (bf16)(__expf(m - e[c]) * zinv);
}

// ---------------------------------------------------------------------------
// sb = relu((1+se) * (gamma*cam + bn_relu(bottle_pre))), written TRANSPOSED
// as sbT[b, n, q] (bf16) so the out-GEMM B-operand is K-contiguous.
// ---------------------------------------------------------------------------
__global__ void k_combine(const float* __restrict__ cam, const float* __restrict__ pre,
                          const float* __restrict__ mean, const float* __restrict__ inv,
                          const float* __restrict__ scale, const float* __restrict__ bias,
                          const float* __restrict__ se, const float* __restrict__ gamma,
                          bf16* __restrict__ sbT)
{
    __shared__ bf16 tile[32][33];
    const int b  = blockIdx.z;
    const int q0 = blockIdx.y * 32;
    const int n0 = blockIdx.x * 32;
    const int tx = threadIdx.x, ty = threadIdx.y;
    const float g = gamma[0];
#pragma unroll
    for (int i = 0; i < 4; ++i) {
        int q = q0 + ty + i * 8;
        size_t idx = ((size_t)b * Q_ + q) * (size_t)HW_ + n0 + tx;
        float bn  = (pre[idx] - mean[q]) * inv[q] * scale[q] + bias[q];
        bn = fmaxf(bn, 0.f);
        float bot = g * cam[idx] + bn;
        float sb  = fmaxf((1.f + se[b * Q_ + q]) * bot, 0.f);
        tile[ty + i * 8][tx] = (bf16)sb;
    }
    __syncthreads();
#pragma unroll
    for (int i = 0; i < 4; ++i) {
        int n = n0 + ty + i * 8;
        sbT[((size_t)b * HW_ + n) * (size_t)Q_ + q0 + tx] = tile[tx][ty + i * 8];
    }
}

// bn2 + relu -> final output (f32)
__global__ void k_bn2(const float* __restrict__ pre,
                      const float* __restrict__ mean, const float* __restrict__ inv,
                      const float* __restrict__ scale, const float* __restrict__ bias,
                      float* __restrict__ out)
{
    const size_t total = (size_t)B_ * O_ * HW_;
    for (size_t i = blockIdx.x * (size_t)blockDim.x + threadIdx.x; i < total;
         i += (size_t)gridDim.x * blockDim.x) {
        int o = (int)((i / HW_) % O_);
        float v = (pre[i] - mean[o]) * inv[o] * scale[o] + bias[o];
        out[i] = fmaxf(v, 0.f);
    }
}

// ---------------------------------------------------------------------------
// Host launcher
// ---------------------------------------------------------------------------
extern "C" void kernel_launch(void* const* d_in, const int* in_sizes, int n_in,
                              void* d_out, int out_size, void* d_ws, size_t ws_size,
                              hipStream_t stream)
{
    (void)in_sizes; (void)n_in; (void)out_size; (void)ws_size;
    const float* x     = (const float*)d_in[0];
    const float* Wproj = (const float*)d_in[1];
    const float* bn1s  = (const float*)d_in[2];
    const float* bn1b  = (const float*)d_in[3];
    const float* Wse1  = (const float*)d_in[4];
    const float* bse1  = (const float*)d_in[5];
    const float* Wse2  = (const float*)d_in[6];
    const float* bse2  = (const float*)d_in[7];
    const float* gamma = (const float*)d_in[8];
    const float* Wout  = (const float*)d_in[9];
    const float* bn2s  = (const float*)d_in[10];
    const float* bn2b  = (const float*)d_in[11];
    float* out = (float*)d_out;

    // workspace carve-up (256B aligned)
    char*  ws  = (char*)d_ws;
    size_t off = 0;
    auto alloc = [&](size_t bytes) -> char* {
        char* p = ws + off;
        off = (off + bytes + 255) & ~(size_t)255;
        return p;
    };
    const size_t XN = (size_t)B_ * C_ * HW_;
    bf16*  xbf    = (bf16*)alloc(XN * 2);                       // x   [B,C,N] bf16
    bf16*  xtbf   = (bf16*)alloc(XN * 2);                       // x^T [B,N,C] bf16
    bf16*  wpbf   = (bf16*)alloc((size_t)Q_ * C_ * 2);          // W_proj bf16
    bf16*  wobf   = (bf16*)alloc((size_t)O_ * Q_ * 2);          // W_out  bf16
    bf16*  qfbf   = (bf16*)alloc((size_t)B_ * Q_ * HW_ * 2);    // bn_relu(bottle) bf16
    bf16*  attnbf = (bf16*)alloc((size_t)B_ * Q_ * C_ * 2);     // attn bf16
    bf16*  sbtbf  = (bf16*)alloc((size_t)B_ * HW_ * Q_ * 2);    // sb^T [B,N,Q] bf16
    float* bottle = (float*)alloc((size_t)B_ * Q_ * HW_ * 4);   // proj pre-BN f32
    float* energy = (float*)alloc((size_t)B_ * Q_ * C_ * 4);    // energy f32
    float* pooled = (float*)alloc((size_t)B_ * C_ * 4);
    float* se     = (float*)alloc((size_t)B_ * Q_ * 4);
    float* bn1m   = (float*)alloc((size_t)Q_ * 4);
    float* bn1i   = (float*)alloc((size_t)Q_ * 4);
    float* bn2m   = (float*)alloc((size_t)O_ * 4);
    float* bn2i   = (float*)alloc((size_t)O_ * 4);
    // safe aliases (consumers/producers strictly ordered):
    float* cam    = (float*)xbf;   // x_bf16 dead after energy GEMM
    float* outpre = (float*)xtbf;  // xT_bf16 dead after cam GEMM

    // 1) convert + transpose x; convert weights; pooled; SE gate
    k_cvt_transpose<<<dim3(HW_ / 32, C_ / 32, B_), dim3(32, 8), 0, stream>>>(x, xbf, xtbf);
    k_cvt<<<(Q_ * C_ + 255) / 256, 256, 0, stream>>>(Wproj, wpbf, Q_ * C_);
    k_cvt<<<(O_ * Q_ + 255) / 256, 256, 0, stream>>>(Wout, wobf, O_ * Q_);
    k_pooled<<<B_ * C_, 256, 0, stream>>>(x, pooled);
    k_se<<<B_, 256, 0, stream>>>(pooled, Wse1, bse1, Wse2, bse2, se);

    // 2) proj GEMM: bottle_pre[b] = W_proj x[b]   (M=Q, N=HW, K=C)
    k_gemm_bf16<<<dim3(HW_ / 128, Q_ / 128, B_), 256, 0, stream>>>(
        wpbf, C_, 0,
        xtbf, C_, (long long)HW_ * C_,
        bottle, HW_, (long long)Q_ * HW_,
        Q_, HW_, C_);
    k_bn_stats<<<Q_, 256, 0, stream>>>(bottle, Q_, bn1m, bn1i);
    k_bn1_qf<<<4096, 256, 0, stream>>>(bottle, bn1m, bn1i, bn1s, bn1b, qfbf);

    // 3) energy GEMM: energy[b] = qf[b] x[b]^T    (M=Q, N=C, K=HW)
    k_gemm_bf16<<<dim3(C_ / 128, Q_ / 128, B_), 256, 0, stream>>>(
        qfbf, HW_, (long long)Q_ * HW_,
        xbf, HW_, (long long)C_ * HW_,
        energy, C_, (long long)Q_ * C_,
        Q_, C_, HW_);
    k_softmax<<<B_ * Q_, 256, 0, stream>>>(energy, attnbf);

    // 4) cam GEMM: cam[b] = attn[b] x[b]          (M=Q, N=HW, K=C)
    k_gemm_bf16<<<dim3(HW_ / 128, Q_ / 128, B_), 256, 0, stream>>>(
        attnbf, C_, (long long)Q_ * C_,
        xtbf, C_, (long long)HW_ * C_,
        cam, HW_, (long long)Q_ * HW_,
        Q_, HW_, C_);
    k_combine<<<dim3(HW_ / 32, Q_ / 32, B_), dim3(32, 8), 0, stream>>>(
        cam, bottle, bn1m, bn1i, bn1s, bn1b, se, gamma, sbtbf);

    // 5) out GEMM: out_pre[b] = W_out sb[b]       (M=O, N=HW, K=Q)
    k_gemm_bf16<<<dim3(HW_ / 128, O_ / 128, B_), 256, 0, stream>>>(
        wobf, Q_, 0,
        sbtbf, Q_, (long long)HW_ * Q_,
        outpre, HW_, (long long)O_ * HW_,
        O_, HW_, Q_);
    k_bn_stats<<<O_, 256, 0, stream>>>(outpre, O_, bn2m, bn2i);
    k_bn2<<<4096, 256, 0, stream>>>(outpre, bn2m, bn2i, bn2s, bn2b, out);
}
